// GraphAttentionModule_39968965657003
// MI455X (gfx1250) — matrix operand order repaired and checked by reading, not compiled, not run on hardware
//
#include <hip/hip_runtime.h>
#include <hip/hip_bf16.h>

// -------------------------------------------------------------------------
// GAT layer for MI455X (gfx1250, wave32, WMMA).
//
// Math:  Wh = X @ W per head;  e_ij = leakyrelu(fs_i + fd_j);  mask by adj;
//        softmax over j;  out = attn @ Wh, heads concatenated.
// Trick: leakyrelu is monotonic => row max m_i = leakyrelu(fs_i + max_j fd_j)
//        over the neighbor set, computable before the accumulation pass, so
//        no online-softmax rescaling is needed.
//
// WMMA layouts used here (wave32; C per CDNA5 ISA ch 7.12.2; A/B use contiguous k, not ISA order):
//  A f16 16x32:  lane<16: row=lane, K half i -> K = i
//                lane>=16: row=lane-16,       K = 16 + i
//  B f16 32x16:  lane<16: col=lane,   halves i -> K = i
//                lane>=16: col=lane-16,          K = 16 + i
//  C f32 16x16:  vgpr r: lane<16 -> (row=r,  col=lane)
//                        lane>=16 -> (row=r+8, col=lane-16)
// Producers write the pre-swizzled B layout so consumers do contiguous loads.
// -------------------------------------------------------------------------

typedef __attribute__((ext_vector_type(16))) _Float16 v16h;
typedef __attribute__((ext_vector_type(8)))  _Float16 v8h;
typedef __attribute__((ext_vector_type(8)))  float    v8f;
typedef __attribute__((ext_vector_type(4)))  float    fx4;
typedef __attribute__((ext_vector_type(4)))  int      ix4;

#define NN   4096   // nodes
#define FIN  512    // input features
#define NH   8      // heads
#define FO   64     // out features per head

#define WMMA_F16(A, B, C) \
  __builtin_amdgcn_wmma_f32_16x16x32_f16(false, (A), false, (B), (short)0, (C), false, false)

// ---------------------------------------------------------------------------
// Kernel 1: swizzle W [H][512][64] f32 -> f16 B-fragment layout.
// tile index (h, kt in 0..15 of K=32, u in 0..3 of 16 cols): 512 halves/tile.
// ---------------------------------------------------------------------------
__global__ void gat_prep_w(const float* __restrict__ W, _Float16* __restrict__ Wswz) {
  int tid  = blockIdx.x * blockDim.x + threadIdx.x;
  int lane = tid & 31;
  int u    = (tid >> 5) & 3;
  int kt   = (tid >> 7) & 15;
  int h    = tid >> 11;
  if (h >= NH) return;
  int col   = u * 16 + (lane & 15);
  int kbase = kt * 32 + ((lane >= 16) ? 16 : 0);
  v16h o;
#pragma unroll
  for (int i = 0; i < 16; ++i)
    o[i] = (_Float16)W[(h * FIN + kbase + i) * FO + col];
  *(v16h*)(Wswz + (((h * 16 + kt) * 4 + u) * 512) + lane * 16) = o;
}

// ---------------------------------------------------------------------------
// Kernel 2: projection GEMM (one wave per block; 16 rows x 64 cols, K=512).
// Stores Wh directly in the phase-2 B-swizzled f16 layout and computes
// f_src/f_dst (dot of Wh rows with a1/a2) in the epilogue.
// ---------------------------------------------------------------------------
__global__ void __launch_bounds__(32) gat_proj(
    const float* __restrict__ X, const _Float16* __restrict__ Wswz,
    const float* __restrict__ a, _Float16* __restrict__ Bswz,
    float* __restrict__ f_src, float* __restrict__ f_dst) {
  int mtile = blockIdx.x;           // 0..255 (16 rows each)
  int h     = blockIdx.y;
  int lane  = threadIdx.x;
  int rowA  = lane & 15;
  int hi    = (lane >= 16) ? 1 : 0; // lane-half selects K sub-range
  int n     = mtile * 16 + rowA;

  const float* xrow = X + (size_t)n * FIN + hi * 16;
  v8f acc0 = {}, acc1 = {}, acc2 = {}, acc3 = {};

  for (int kt = 0; kt < 16; ++kt) {
    fx4 x0 = *(const fx4*)(xrow + kt * 32 + 0);
    fx4 x1 = *(const fx4*)(xrow + kt * 32 + 4);
    fx4 x2 = *(const fx4*)(xrow + kt * 32 + 8);
    fx4 x3 = *(const fx4*)(xrow + kt * 32 + 12);
    v16h A;
    A[0]=(_Float16)x0.x;  A[1]=(_Float16)x0.y;  A[2]=(_Float16)x0.z;  A[3]=(_Float16)x0.w;
    A[4]=(_Float16)x1.x;  A[5]=(_Float16)x1.y;  A[6]=(_Float16)x1.z;  A[7]=(_Float16)x1.w;
    A[8]=(_Float16)x2.x;  A[9]=(_Float16)x2.y;  A[10]=(_Float16)x2.z; A[11]=(_Float16)x2.w;
    A[12]=(_Float16)x3.x; A[13]=(_Float16)x3.y; A[14]=(_Float16)x3.z; A[15]=(_Float16)x3.w;

    const _Float16* bp = Wswz + ((h * 16 + kt) * 4) * 512 + lane * 16;
    v16h B0 = *(const v16h*)(bp);
    v16h B1 = *(const v16h*)(bp + 512);
    v16h B2 = *(const v16h*)(bp + 1024);
    v16h B3 = *(const v16h*)(bp + 1536);
    acc0 = WMMA_F16(A, B0, acc0);
    acc1 = WMMA_F16(A, B1, acc1);
    acc2 = WMMA_F16(A, B2, acc2);
    acc3 = WMMA_F16(A, B3, acc3);
  }

  // --- store Wh into the B-swizzled layout used by the attention kernel ---
  // element (k = n%32, col) lives at lane_t = col + (k>=16)*16, half i = k&15.
  int jt     = mtile >> 1;
  int lane_t = (lane & 15) + (mtile & 1) * 16;
  int ibase  = hi * 8;                       // rows r+hi*8 -> halves ibase+r
#define STORE_U(u, acc)                                                         \
  { v8h t;                                                                      \
    _Pragma("unroll") for (int r = 0; r < 8; ++r) t[r] = (_Float16)(acc)[r];    \
    *(v8h*)(Bswz + (((h * 128 + jt) * 4 + (u)) * 512) + lane_t * 16 + ibase) = t; }
  STORE_U(0, acc0) STORE_U(1, acc1) STORE_U(2, acc2) STORE_U(3, acc3)
#undef STORE_U

  // --- epilogue: f_src/f_dst = Wh . a1 / a2 ---------------------------------
  __shared__ float smp[32][8];
  float as1[4], as2[4];
#pragma unroll
  for (int u = 0; u < 4; ++u) {
    as1[u] = a[h * 128 + u * 16 + (lane & 15)];
    as2[u] = a[h * 128 + 64 + u * 16 + (lane & 15)];
  }
  float ps[8], pd[8];
#pragma unroll
  for (int r = 0; r < 8; ++r) {
    ps[r] = acc0[r] * as1[0] + acc1[r] * as1[1] + acc2[r] * as1[2] + acc3[r] * as1[3];
    pd[r] = acc0[r] * as2[0] + acc1[r] * as2[1] + acc2[r] * as2[2] + acc3[r] * as2[3];
  }
#pragma unroll
  for (int r = 0; r < 8; ++r) smp[lane][r] = ps[r];
  __syncthreads();
  if (lane < 16) {
    int lo = (lane < 8) ? 0 : 16, r = lane & 7;
    float s = 0.0f;
#pragma unroll
    for (int l = 0; l < 16; ++l) s += smp[lo + l][r];
    f_src[h * NN + mtile * 16 + lane] = s;
  }
  __syncthreads();
#pragma unroll
  for (int r = 0; r < 8; ++r) smp[lane][r] = pd[r];
  __syncthreads();
  if (lane < 16) {
    int lo = (lane < 8) ? 0 : 16, r = lane & 7;
    float s = 0.0f;
#pragma unroll
    for (int l = 0; l < 16; ++l) s += smp[lo + l][r];
    f_dst[h * NN + mtile * 16 + lane] = s;
  }
}

// ---------------------------------------------------------------------------
// Kernel 3: fused masked-softmax attention + aggregation.
// Block = (head, 16-row tile), 8 waves; each wave streams 32-wide j tiles.
// ---------------------------------------------------------------------------
__global__ void __launch_bounds__(256) gat_attn(
    const int* __restrict__ adj, const float* __restrict__ f_src,
    const float* __restrict__ f_dst, const _Float16* __restrict__ Bswz,
    float* __restrict__ out) {
  int rt  = blockIdx.x;   // row tile 0..255
  int h   = blockIdx.y;
  int tid = threadIdx.x;

  __shared__ float sm_red[256];
  __shared__ float sm_m[16];
  __shared__ float sm_fs[16];
  __shared__ float sm_l[16];
  __shared__ float sm_acc[16 * 64];

#pragma unroll
  for (int k = 0; k < 4; ++k) sm_acc[tid + k * 256] = 0.0f;

  // --- phase 1: m_i = leakyrelu(fs_i + max_{adj} fd_j) ----------------------
  {
    int row = tid >> 4, sl = tid & 15;
    const int*   arow = adj + (size_t)(rt * 16 + row) * NN;
    const float* fd   = f_dst + h * NN;
    float mx = -3.0e38f;
    for (int it = 0; it < 64; ++it) {
      int j = sl * 4 + it * 64;
      ix4 av = *(const ix4*)(arow + j);
      fx4 fv = *(const fx4*)(fd + j);
      if (av.x > 0) mx = fmaxf(mx, fv.x);
      if (av.y > 0) mx = fmaxf(mx, fv.y);
      if (av.z > 0) mx = fmaxf(mx, fv.z);
      if (av.w > 0) mx = fmaxf(mx, fv.w);
    }
    sm_red[tid] = mx;
  }
  __syncthreads();
  if (tid < 16) {
    float mx = -3.0e38f;
#pragma unroll
    for (int s = 0; s < 16; ++s) mx = fmaxf(mx, sm_red[tid * 16 + s]);
    float fs = f_src[h * NN + rt * 16 + tid];
    float s0 = fs + mx;                       // self-loop guarantees a neighbor
    sm_m[tid]  = (s0 > 0.0f) ? s0 : 0.2f * s0;
    sm_fs[tid] = fs;
    sm_l[tid]  = 0.0f;
  }
  __syncthreads();

  // --- phase 2: acc += exp(e - m) * Wh via WMMA -----------------------------
  int wave = tid >> 5, lane = tid & 31;
  int rowA = lane & 15;
  int hi   = (lane >= 16) ? 1 : 0;
  int c    = hi * 16;
  float m_row  = sm_m[rowA];
  float fs_row = sm_fs[rowA];
  const int*   arow = adj + (size_t)(rt * 16 + rowA) * NN + c;
  const float* fdp  = f_dst + h * NN + c;

  v8f acc0 = {}, acc1 = {}, acc2 = {}, acc3 = {};
  float lsum = 0.0f;

  for (int t = wave; t < 128; t += 8) {
    int jb = t * 32;
    if (t + 8 < 128) __builtin_prefetch(arow + (t + 8) * 32, 0, 1);  // global_prefetch_b8
    ix4 a0 = *(const ix4*)(arow + jb);
    ix4 a1 = *(const ix4*)(arow + jb + 4);
    ix4 a2 = *(const ix4*)(arow + jb + 8);
    ix4 a3 = *(const ix4*)(arow + jb + 12);
    fx4 f0 = *(const fx4*)(fdp + jb);
    fx4 f1 = *(const fx4*)(fdp + jb + 4);
    fx4 f2 = *(const fx4*)(fdp + jb + 8);
    fx4 f3 = *(const fx4*)(fdp + jb + 12);

    v16h A;
#define PEL(idx, av, fv)                                                       \
  { float e = fs_row + (fv); e = (e > 0.0f) ? e : 0.2f * e;                    \
    float p = ((av) > 0)                                                       \
        ? __builtin_amdgcn_exp2f((e - m_row) * 1.44269504088896f) : 0.0f;      \
    lsum += p; A[idx] = (_Float16)p; }
    PEL(0,  a0.x, f0.x) PEL(1,  a0.y, f0.y) PEL(2,  a0.z, f0.z) PEL(3,  a0.w, f0.w)
    PEL(4,  a1.x, f1.x) PEL(5,  a1.y, f1.y) PEL(6,  a1.z, f1.z) PEL(7,  a1.w, f1.w)
    PEL(8,  a2.x, f2.x) PEL(9,  a2.y, f2.y) PEL(10, a2.z, f2.z) PEL(11, a2.w, f2.w)
    PEL(12, a3.x, f3.x) PEL(13, a3.y, f3.y) PEL(14, a3.z, f3.z) PEL(15, a3.w, f3.w)
#undef PEL

    const _Float16* bp = Bswz + ((h * 128 + t) * 4) * 512 + lane * 16;
    v16h B0 = *(const v16h*)(bp);
    v16h B1 = *(const v16h*)(bp + 512);
    v16h B2 = *(const v16h*)(bp + 1024);
    v16h B3 = *(const v16h*)(bp + 1536);
    acc0 = WMMA_F16(A, B0, acc0);
    acc1 = WMMA_F16(A, B1, acc1);
    acc2 = WMMA_F16(A, B2, acc2);
    acc3 = WMMA_F16(A, B3, acc3);
  }

  // --- cross-wave reduction in LDS (ds_add_f32) -----------------------------
  atomicAdd(&sm_l[rowA], lsum);
  int colb = lane & 15;
#pragma unroll
  for (int r = 0; r < 8; ++r) {
    int row = r + hi * 8;
    atomicAdd(&sm_acc[row * 64 +  0 + colb], acc0[r]);
    atomicAdd(&sm_acc[row * 64 + 16 + colb], acc1[r]);
    atomicAdd(&sm_acc[row * 64 + 32 + colb], acc2[r]);
    atomicAdd(&sm_acc[row * 64 + 48 + colb], acc3[r]);
  }
  __syncthreads();

  // --- normalize and write out[n][h*64+col] ---------------------------------
  {
    int e0 = tid * 4;
    int row = e0 >> 6, col = e0 & 63;
    float inv = 1.0f / sm_l[row];
    fx4 v = *(fx4*)&sm_acc[e0];
    v.x *= inv; v.y *= inv; v.z *= inv; v.w *= inv;
    *(fx4*)(out + (size_t)(rt * 16 + row) * (NH * FO) + h * FO + col) = v;
  }
}

// ---------------------------------------------------------------------------
extern "C" void kernel_launch(void* const* d_in, const int* in_sizes, int n_in,
                              void* d_out, int out_size, void* d_ws, size_t ws_size,
                              hipStream_t stream) {
  const float* X   = (const float*)d_in[0];   // [4096, 512]
  const int*   adj = (const int*)d_in[1];     // [4096, 4096]
  const float* W   = (const float*)d_in[2];   // [8, 512, 64]
  const float* a   = (const float*)d_in[3];   // [8, 128]
  float* out = (float*)d_out;                 // [4096, 512]

  char* ws = (char*)d_ws;
  _Float16* Wswz = (_Float16*)ws;                              // 512 KB
  _Float16* Bswz = (_Float16*)(ws + 512 * 1024);               // 4 MB
  float* f_src   = (float*)(ws + 512 * 1024 + 4 * 1024 * 1024);// 128 KB
  float* f_dst   = f_src + NH * NN;                            // 128 KB

  gat_prep_w<<<64, 256, 0, stream>>>(W, Wswz);
  gat_proj<<<dim3(256, 8), 32, 0, stream>>>(X, Wswz, a, Bswz, f_src, f_dst);
  gat_attn<<<dim3(256, 8), 256, 0, stream>>>(adj, f_src, f_dst, Bswz, out);
}
